// TreeDecoder_74217034875211
// MI455X (gfx1250) — compile-verified, matrix operand order checked by reading, not run on hardware
//
#include <hip/hip_runtime.h>
#include <hip/hip_bf16.h>
#include <math.h>

typedef __bf16 bf16;
typedef __attribute__((ext_vector_type(16))) __bf16 v16bf;
typedef __attribute__((ext_vector_type(8)))  __bf16 v8bf;
typedef __attribute__((ext_vector_type(8)))  float  v8f;
typedef long long ll;

#define Hh    256
#define Vv    1024
#define BATCH 32
#define LSEQ  1024
#define TDEPTH 9
#define NTOT  1023
#define NMAX  512

// ---------------------------------------------------------------------------
// Fragment load: 16x32 bf16 tile, row-major with K contiguous (ISA 7.12.2
// 16-bit A layout; B operand of A@Wt^T has the identical layout).
// rowp already points at row base + (lane>=16 ? 8 : 0) elements.
// elements 0..7 <- K = k..k+7 ; elements 8..15 <- K = k+16..k+23
// ---------------------------------------------------------------------------
__device__ __forceinline__ v16bf load_frag(const bf16* rowp, int k) {
    v8bf a = *(const v8bf*)(rowp + k);
    v8bf b = *(const v8bf*)(rowp + k + 16);
    v16bf r;
#pragma unroll
    for (int i = 0; i < 8; ++i) { r[i] = a[i]; r[i + 8] = b[i]; }
    return r;
}

// ---------------------------------------------------------------------------
// Generic batched WMMA GEMM:
//   C[b] = act( A1[b](M,K1) @ W1[b](N,K1)^T  (+ A2[b](M,K2) @ W2[b](N,K2)^T)
//               (+ bias1) (+ bias2) )
// One wave per block; each wave computes a 32(M) x 64(N) tile with 8
// v_wmma_f32_16x16x32_bf16 accumulators. The two A fragments are reused
// across 4 N-tiles, and each W fragment is reused across both M-subtiles:
// 12 b128 loads per 8 WMMAs in the k-loop (1.5 loads/WMMA).
// Out-of-range rows/cols are clamped on load and guarded on store.
// ---------------------------------------------------------------------------
__global__ __launch_bounds__(32) void wmma_gemm(
    const bf16* __restrict__ A1, ll sA1, int lda1,
    const bf16* __restrict__ W1, ll sW1, int ldw1,
    const bf16* __restrict__ A2, ll sA2, int lda2,
    const bf16* __restrict__ W2, ll sW2, int ldw2,
    const float* __restrict__ bias1, const float* __restrict__ bias2,
    float* __restrict__ Cf, ll sCf, int ldcf,
    bf16* __restrict__ Cb, ll sCb, int ldcb,
    int M, int Ncols, int K1, int K2, int act)
{
    const int lane = threadIdx.x;
    const int hi = lane >> 4;       // which K-half of the fragment
    const int lo = lane & 15;       // row within tile
    const int m0 = blockIdx.x * 32;
    const int n0 = blockIdx.y * 64;
    const int b  = blockIdx.z;

    v8f acc[8] = {};                // acc[t] = (m0..m0+15, n-tile t); acc[4+t] = (m0+16..m0+31)

    auto mm = [&](const bf16* A, ll sA, int lda,
                  const bf16* W, ll sW, int ldw, int K) {
        const bf16* Ab = A + (ll)b * sA;
        const bf16* Wb = W + (ll)b * sW;
        int ar0 = m0 + lo;      if (ar0 > M - 1) ar0 = M - 1;
        int ar1 = m0 + 16 + lo; if (ar1 > M - 1) ar1 = M - 1;
        const bf16* ap0 = Ab + (ll)ar0 * lda + hi * 8;
        const bf16* ap1 = Ab + (ll)ar1 * lda + hi * 8;
        const bf16* wp[4];
#pragma unroll
        for (int t = 0; t < 4; ++t) {
            int wr = n0 + t * 16 + lo; if (wr > Ncols - 1) wr = Ncols - 1;
            wp[t] = Wb + (ll)wr * ldw + hi * 8;
        }
        for (int k = 0; k < K; k += 32) {
            v16bf af0 = load_frag(ap0, k);
            v16bf af1 = load_frag(ap1, k);
#pragma unroll
            for (int t = 0; t < 4; ++t) {
                v16bf wf = load_frag(wp[t], k);
                acc[t] = __builtin_amdgcn_wmma_f32_16x16x32_bf16(
                    false, af0, false, wf, (short)0, acc[t], false, false);
                acc[4 + t] = __builtin_amdgcn_wmma_f32_16x16x32_bf16(
                    false, af1, false, wf, (short)0, acc[4 + t], false, false);
            }
        }
    };

    mm(A1, sA1, lda1, W1, sW1, ldw1, K1);
    if (A2) mm(A2, sA2, lda2, W2, sW2, ldw2, K2);

    float* cf = Cf ? Cf + (ll)b * sCf : nullptr;
    bf16*  cb = Cb ? Cb + (ll)b * sCb : nullptr;
#pragma unroll
    for (int u = 0; u < 2; ++u) {
#pragma unroll
        for (int t = 0; t < 4; ++t) {
#pragma unroll
            for (int r = 0; r < 8; ++r) {
                int row = m0 + u * 16 + r + hi * 8;   // C layout: lanes>=16 hold M+8
                int col = n0 + t * 16 + lo;
                if (row < M && col < Ncols) {
                    float v = acc[u * 4 + t][r];
                    if (bias1) v += bias1[col];
                    if (bias2) v += bias2[col];
                    if (act)   v = tanhf(v);
                    if (cf) cf[(ll)row * ldcf + col] = v;
                    if (cb) cb[(ll)row * ldcb + col] = (bf16)v;
                }
            }
        }
    }
}

// ---------------------------------------------------------------------------
// softmax over L of t = exp(s) (reference does softmax(exp(scores))).
// scores layout: (B, L, NMAX) f32, column j. Output transposed bf16 (B,NMAX,L)
// so the context GEMM consumes it as a row-major A operand.
// ---------------------------------------------------------------------------
__global__ void softmax_kernel(const float* __restrict__ scores,
                               bf16* __restrict__ wsmT, int n)
{
    int j = blockIdx.x, b = blockIdx.y, t = threadIdx.x;
    const float* s = scores + (ll)b * LSEQ * NMAX + j;
    __shared__ float red[256];
    float m = -1e30f;
    for (int l = t; l < LSEQ; l += 256) m = fmaxf(m, expf(s[(ll)l * NMAX]));
    red[t] = m; __syncthreads();
    for (int o = 128; o > 0; o >>= 1) { if (t < o) red[t] = fmaxf(red[t], red[t + o]); __syncthreads(); }
    m = red[0]; __syncthreads();
    float sum = 0.f;
    for (int l = t; l < LSEQ; l += 256) sum += expf(expf(s[(ll)l * NMAX]) - m);
    red[t] = sum; __syncthreads();
    for (int o = 128; o > 0; o >>= 1) { if (t < o) red[t] += red[t + o]; __syncthreads(); }
    float inv = 1.0f / red[0];
    bf16* out = wsmT + ((ll)b * NMAX + j) * LSEQ;
    for (int l = t; l < LSEQ; l += 256)
        out[l] = (bf16)(expf(expf(s[(ll)l * NMAX]) - m) * inv);
}

// ---------------------------------------------------------------------------
// Per-node cross-entropy + argmax over V.
// ---------------------------------------------------------------------------
__global__ void loss_argmax_kernel(const float* __restrict__ logits,
                                   const int* __restrict__ tgt,
                                   float* __restrict__ out_pred,
                                   int* __restrict__ pred_cur,
                                   float* __restrict__ loss_acc,
                                   int n, int start)
{
    int j = blockIdx.x, b = blockIdx.y, t = threadIdx.x;
    const float* row = logits + ((ll)b * n + j) * Vv;
    __shared__ float sm[256];
    __shared__ int   si[256];
    float m = -1e30f; int mi = 0;
    for (int v = t; v < Vv; v += 256) { float x = row[v]; if (x > m) { m = x; mi = v; } }
    sm[t] = m; si[t] = mi; __syncthreads();
    for (int o = 128; o > 0; o >>= 1) {
        if (t < o && (sm[t + o] > sm[t] || (sm[t + o] == sm[t] && si[t + o] < si[t]))) {
            sm[t] = sm[t + o]; si[t] = si[t + o];
        }
        __syncthreads();
    }
    m = sm[0]; mi = si[0]; __syncthreads();
    float sum = 0.f;
    for (int v = t; v < Vv; v += 256) sum += expf(row[v] - m);
    sm[t] = sum; __syncthreads();
    for (int o = 128; o > 0; o >>= 1) { if (t < o) sm[t] += sm[t + o]; __syncthreads(); }
    if (t == 0) {
        float lse = m + logf(sm[0]);
        int tg = tgt[b * NTOT + start + j];
        atomicAdd(loss_acc, -(row[tg] - lse));
        out_pred[b * NTOT + start + j] = (float)mi;
        pred_cur[b * NMAX + j] = mi;
    }
}

// x16[b,j,:] = bf16(emb_table[pred[b,j],:])
__global__ void gather_emb_kernel(const float* __restrict__ embtbl,
                                  const int* __restrict__ pred,
                                  bf16* __restrict__ x16, int n)
{
    ll i = (ll)blockIdx.x * blockDim.x + threadIdx.x;
    ll tot = (ll)BATCH * n * Hh;
    if (i >= tot) return;
    int hh = (int)(i % Hh); ll bj = i / Hh;
    int j = (int)(bj % n);  int b = (int)(bj / n);
    int id = pred[b * NMAX + j];
    x16[((ll)b * NMAX + j) * Hh + hh] = (bf16)embtbl[(ll)id * Hh + hh];
}

// LSTM pointwise: gates (i,f,g,o), interleave left/right children.
__global__ void lstm_kernel(const float* __restrict__ gl, const float* __restrict__ gr,
                            const float* __restrict__ c_cur,
                            float* __restrict__ c_next, bf16* __restrict__ h16_next, int n)
{
    int j = blockIdx.x, b = blockIdx.y, hh = threadIdx.x;   // blockDim = H
    ll gi = ((ll)b * n + j) * 4 * Hh;
    float il = gl[gi + hh],  fl = gl[gi + Hh + hh];
    float ggl = gl[gi + 2 * Hh + hh], ol = gl[gi + 3 * Hh + hh];
    float ir = gr[gi + hh],  fr = gr[gi + Hh + hh];
    float ggr = gr[gi + 2 * Hh + hh], orr = gr[gi + 3 * Hh + hh];
    float cc = c_cur[((ll)b * NMAX + j) * Hh + hh];
    auto sig = [](float x) { return 1.0f / (1.0f + expf(-x)); };
    float cl = sig(fl) * cc + sig(il) * tanhf(ggl);
    float hl = sig(ol) * tanhf(cl);
    float cr = sig(fr) * cc + sig(ir) * tanhf(ggr);
    float hr = sig(orr) * tanhf(cr);
    ll li = ((ll)b * NMAX + 2 * j) * Hh + hh;
    ll ri = ((ll)b * NMAX + 2 * j + 1) * Hh + hh;
    c_next[li] = cl;          c_next[ri] = cr;
    h16_next[li] = (bf16)hl;  h16_next[ri] = (bf16)hr;
}

__global__ void f32_to_bf16_kernel(const float* __restrict__ src, bf16* __restrict__ dst, ll count) {
    ll i = (ll)blockIdx.x * blockDim.x + threadIdx.x;
    if (i < count) dst[i] = (bf16)src[i];
}
__global__ void zero_f32_kernel(float* p, ll count) {
    ll i = (ll)blockIdx.x * blockDim.x + threadIdx.x;
    if (i < count) p[i] = 0.0f;
}
__global__ void zero_bf16_kernel(bf16* p, ll count) {
    ll i = (ll)blockIdx.x * blockDim.x + threadIdx.x;
    if (i < count) p[i] = (bf16)0.0f;
}
// (B,L,H) f32 -> (B,H,L) bf16
__global__ void transpose_bf16_kernel(const float* __restrict__ in, bf16* __restrict__ out) {
    ll i = (ll)blockIdx.x * blockDim.x + threadIdx.x;
    if (i >= (ll)BATCH * LSEQ * Hh) return;
    int hh = (int)(i % Hh); ll bl = i / Hh;
    int l = (int)(bl % LSEQ); int b = (int)(bl / LSEQ);
    out[((ll)b * Hh + hh) * LSEQ + l] = (bf16)in[i];
}
__global__ void finalize_loss_kernel(const float* loss_acc, float* out) {
    out[0] = loss_acc[0] / (float)(BATCH * NTOT);
}

// ---------------------------------------------------------------------------
extern "C" void kernel_launch(void* const* d_in, const int* in_sizes, int n_in,
                              void* d_out, int out_size, void* d_ws, size_t ws_size,
                              hipStream_t stream)
{
    const float* input_emb = (const float*)d_in[0];
    const int*   target_ids = (const int*)d_in[1];
    const float* w0_w = (const float*)d_in[3];  const float* w0_b = (const float*)d_in[4];
    const float* w1_w = (const float*)d_in[5];  const float* w1_b = (const float*)d_in[6];
    const float* w2_w = (const float*)d_in[7];  const float* w2_b = (const float*)d_in[8];
    const float* wtt_w = (const float*)d_in[9]; const float* wtt_b = (const float*)d_in[10];
    const float* emb_tbl = (const float*)d_in[11];
    const float* wih_l = (const float*)d_in[12]; const float* whh_l = (const float*)d_in[13];
    const float* wih_r = (const float*)d_in[14]; const float* whh_r = (const float*)d_in[15];

    // ---- workspace carve ----
    char* p = (char*)d_ws;
    auto carve = [&](size_t bytes) -> void* {
        void* r = (void*)p; p += (bytes + 255) & ~(size_t)255; return r;
    };
    bf16* emb16   = (bf16*)carve((size_t)BATCH * LSEQ * Hh * 2);   // 16 MB
    bf16* embT16  = (bf16*)carve((size_t)BATCH * Hh * LSEQ * 2);   // 16 MB
    bf16* wsmT16  = (bf16*)carve((size_t)BATCH * NMAX * LSEQ * 2); // 32 MB
    float* region1 = (float*)carve((size_t)BATCH * LSEQ * NMAX * 4); // 64 MB: scores->logits->gates
    float* c_a = (float*)carve((size_t)BATCH * NMAX * Hh * 4);     // 16 MB
    float* c_b = (float*)carve((size_t)BATCH * NMAX * Hh * 4);     // 16 MB
    bf16* h16_a = (bf16*)carve((size_t)BATCH * NMAX * Hh * 2);     // 8 MB
    bf16* h16_b = (bf16*)carve((size_t)BATCH * NMAX * Hh * 2);     // 8 MB
    bf16* htp16 = (bf16*)carve((size_t)BATCH * NMAX * Hh * 2);
    bf16* es16  = (bf16*)carve((size_t)BATCH * NMAX * Hh * 2);
    bf16* et16  = (bf16*)carve((size_t)BATCH * NMAX * Hh * 2);
    bf16* x16   = (bf16*)carve((size_t)BATCH * NMAX * Hh * 2);
    bf16* w0_16 = (bf16*)carve((size_t)Hh * Hh * 2);
    bf16* w1_16 = (bf16*)carve((size_t)Hh * Hh * 2);
    bf16* w2_16 = (bf16*)carve((size_t)Hh * Hh * 2);
    bf16* wtt16 = (bf16*)carve((size_t)Vv * Hh * 2);
    bf16* wihl16 = (bf16*)carve((size_t)4 * Hh * Hh * 2);
    bf16* whhl16 = (bf16*)carve((size_t)4 * Hh * Hh * 2);
    bf16* wihr16 = (bf16*)carve((size_t)4 * Hh * Hh * 2);
    bf16* whhr16 = (bf16*)carve((size_t)4 * Hh * Hh * 2);
    int*  pred_cur = (int*)carve((size_t)BATCH * NMAX * 4);
    float* loss_acc = (float*)carve(256);

    float* scores = region1;
    float* logits = region1;
    float* g_l = region1;
    float* g_r = region1 + (size_t)BATCH * 256 * 4 * Hh;  // second 32 MB half

    auto cvt = [&](const float* s, bf16* d, ll cnt) {
        f32_to_bf16_kernel<<<dim3((unsigned)((cnt + 255) / 256)), 256, 0, stream>>>(s, d, cnt);
    };

    // ---- setup ----
    {
        ll cnt = (ll)BATCH * NMAX * Hh;
        zero_f32_kernel<<<dim3((unsigned)((cnt + 255) / 256)), 256, 0, stream>>>(c_a, cnt);
        zero_bf16_kernel<<<dim3((unsigned)((cnt + 255) / 256)), 256, 0, stream>>>(h16_a, cnt);
        zero_f32_kernel<<<1, 1, 0, stream>>>(loss_acc, 1);
    }
    cvt(input_emb, emb16, (ll)BATCH * LSEQ * Hh);
    {
        ll cnt = (ll)BATCH * LSEQ * Hh;
        transpose_bf16_kernel<<<dim3((unsigned)((cnt + 255) / 256)), 256, 0, stream>>>(input_emb, embT16);
    }
    cvt(w0_w, w0_16, (ll)Hh * Hh);  cvt(w1_w, w1_16, (ll)Hh * Hh);  cvt(w2_w, w2_16, (ll)Hh * Hh);
    cvt(wtt_w, wtt16, (ll)Vv * Hh);
    cvt(wih_l, wihl16, (ll)4 * Hh * Hh); cvt(whh_l, whhl16, (ll)4 * Hh * Hh);
    cvt(wih_r, wihr16, (ll)4 * Hh * Hh); cvt(whh_r, whhr16, (ll)4 * Hh * Hh);

    bf16* h16_cur = h16_a; bf16* h16_nxt = h16_b;
    float* c_cur = c_a;    float* c_nxt = c_b;

    for (int level = 0; level <= TDEPTH; ++level) {
        int n = 1 << level, start = n - 1;
        dim3 gH((n + 31) / 32, Hh / 64, BATCH);

        // htp = h @ w0^T + w0_b            -> htp16 (B,NMAX,H)
        wmma_gemm<<<gH, 32, 0, stream>>>(
            h16_cur, (ll)NMAX * Hh, Hh,  w0_16, 0, Hh,
            nullptr, 0, 0, nullptr, 0, 0,
            w0_b, nullptr,
            nullptr, 0, 0,  htp16, (ll)NMAX * Hh, Hh,
            n, Hh, Hh, 0, 0);

        // scores = emb(L,H) @ htp^T        -> (B,L,NMAX) f32
        dim3 gS(LSEQ / 32, (n + 63) / 64, BATCH);
        wmma_gemm<<<gS, 32, 0, stream>>>(
            emb16, (ll)LSEQ * Hh, Hh,  htp16, (ll)NMAX * Hh, Hh,
            nullptr, 0, 0, nullptr, 0, 0,
            nullptr, nullptr,
            scores, (ll)LSEQ * NMAX, NMAX,  nullptr, 0, 0,
            LSEQ, n, Hh, 0, 0);

        softmax_kernel<<<dim3(n, BATCH), 256, 0, stream>>>(scores, wsmT16, n);

        // e_s = wsmT(n,L) @ embT(H,L)^T    -> es16
        wmma_gemm<<<gH, 32, 0, stream>>>(
            wsmT16, (ll)NMAX * LSEQ, LSEQ,  embT16, (ll)Hh * LSEQ, LSEQ,
            nullptr, 0, 0, nullptr, 0, 0,
            nullptr, nullptr,
            nullptr, 0, 0,  es16, (ll)NMAX * Hh, Hh,
            n, Hh, LSEQ, 0, 0);

        // e_t = tanh(e_s@w1^T + h@w2^T + b1 + b2)   -> et16
        wmma_gemm<<<gH, 32, 0, stream>>>(
            es16, (ll)NMAX * Hh, Hh,  w1_16, 0, Hh,
            h16_cur, (ll)NMAX * Hh, Hh,  w2_16, 0, Hh,
            w1_b, w2_b,
            nullptr, 0, 0,  et16, (ll)NMAX * Hh, Hh,
            n, Hh, Hh, Hh, 1);

        // logits = e_t @ wtt^T + wtt_b     -> (B,n,V) f32
        dim3 gL((n + 31) / 32, Vv / 64, BATCH);
        wmma_gemm<<<gL, 32, 0, stream>>>(
            et16, (ll)NMAX * Hh, Hh,  wtt16, 0, Hh,
            nullptr, 0, 0, nullptr, 0, 0,
            wtt_b, nullptr,
            logits, (ll)n * Vv, Vv,  nullptr, 0, 0,
            n, Vv, Hh, 0, 0);

        loss_argmax_kernel<<<dim3(n, BATCH), 256, 0, stream>>>(
            logits, target_ids, (float*)d_out, pred_cur, loss_acc, n, start);

        if (level < TDEPTH) {
            ll tot = (ll)BATCH * n * Hh;
            gather_emb_kernel<<<dim3((unsigned)((tot + 255) / 256)), 256, 0, stream>>>(
                emb_tbl, pred_cur, x16, n);

            dim3 gG((n + 31) / 32, (4 * Hh) / 64, BATCH);
            wmma_gemm<<<gG, 32, 0, stream>>>(
                x16, (ll)NMAX * Hh, Hh,  wihl16, 0, Hh,
                h16_cur, (ll)NMAX * Hh, Hh,  whhl16, 0, Hh,
                nullptr, nullptr,
                g_l, (ll)n * 4 * Hh, 4 * Hh,  nullptr, 0, 0,
                n, 4 * Hh, Hh, Hh, 0);
            wmma_gemm<<<gG, 32, 0, stream>>>(
                x16, (ll)NMAX * Hh, Hh,  wihr16, 0, Hh,
                h16_cur, (ll)NMAX * Hh, Hh,  whhr16, 0, Hh,
                nullptr, nullptr,
                g_r, (ll)n * 4 * Hh, 4 * Hh,  nullptr, 0, 0,
                n, 4 * Hh, Hh, Hh, 0);

            lstm_kernel<<<dim3(n, BATCH), Hh, 0, stream>>>(g_l, g_r, c_cur, c_nxt, h16_nxt, n);

            bf16* th = h16_cur; h16_cur = h16_nxt; h16_nxt = th;
            float* tc = c_cur;  c_cur = c_nxt;     c_nxt = tc;
        }
    }

    finalize_loss_kernel<<<1, 1, 0, stream>>>(loss_acc, (float*)d_out + (size_t)BATCH * NTOT);
}
